// MultiHeadAttention_64922725646572
// MI455X (gfx1250) — matrix operand order repaired and checked by reading, not compiled, not run on hardware
//
#include <hip/hip_runtime.h>

typedef _Float16 f16;
typedef __attribute__((ext_vector_type(16))) _Float16 v16h;
typedef __attribute__((ext_vector_type(8)))  float    v8f;

#define B_   4
#define S_   2048
#define D_   1024
#define H_   16
#define HD_  64
#define NQKV 3072
#define MROWS (B_ * S_)   // 8192

// ---------------------------------------------------------------------------
// WMMA fragment loaders (ISA 7.12.2 layouts), each = two 16B loads.
// A-matrix 16x32 f16: lane r=lane&15 holds row M=r; hi=lane>>4 selects
//   halfs K = hi*8..hi*8+7 (chunk0) and K = 16+hi*8.. (chunk1).
// B-matrix 32x16 f16: lane holds column N=r; same K split as A (hi*8.., 16+hi*8..).
// Both assume the source is "K-contiguous" row-major (A row-major, Bt row-major).
// ---------------------------------------------------------------------------
static __device__ __forceinline__ v16h load_frag_a(const f16* base, int strideHalfs, int lane) {
  const int r = lane & 15, hi = lane >> 4;
  const f16* p = base + (size_t)r * strideHalfs + hi * 8;
  uint4 c0 = *(const uint4*)(p);
  uint4 c1 = *(const uint4*)(p + 16);
  v16h a; ((uint4*)&a)[0] = c0; ((uint4*)&a)[1] = c1; return a;
}
static __device__ __forceinline__ v16h load_frag_b(const f16* base, int strideHalfs, int lane) {
  const int r = lane & 15, hi = lane >> 4;
  const f16* p = base + (size_t)r * strideHalfs + hi * 8;
  uint4 c0 = *(const uint4*)(p);
  uint4 c1 = *(const uint4*)(p + 16);
  v16h a; ((uint4*)&a)[0] = c0; ((uint4*)&a)[1] = c1; return a;
}

// Async global -> LDS copy of one 16B chunk (ASYNCcnt-tracked, no VGPR round trip).
// LDS address = low 32 bits of the generic shared pointer (aperture mapping).
static __device__ __forceinline__ void async_copy_b128(const f16* gsrc, f16* ldst) {
  unsigned lds_addr = (unsigned)(uintptr_t)ldst;
  unsigned long long gaddr = (unsigned long long)(uintptr_t)gsrc;
  asm volatile("global_load_async_to_lds_b128 %0, %1, off"
               :: "v"(lds_addr), "v"(gaddr) : "memory");
}

// ---------------------------------------------------------------------------
// f32 -> f16 convert (x), 4 elements / thread
// ---------------------------------------------------------------------------
__global__ __launch_bounds__(256) void cvt_f32_f16_kernel(const float* __restrict__ in,
                                                          f16* __restrict__ out, int n) {
  int idx = (blockIdx.x * 256 + threadIdx.x) * 4;
  if (idx < n) {
    float4 v = *(const float4*)(in + idx);
    f16 t[4] = { (f16)v.x, (f16)v.y, (f16)v.z, (f16)v.w };
    *(uint2*)(out + idx) = *(uint2*)t;
  }
}

// ---------------------------------------------------------------------------
// Transposing convert: W[R][C] f32 (row-major) -> Wt[C][R] f16 (row-major).
// Gives B-operand in K-contiguous (Bt) layout for the WMMA GEMMs.
// ---------------------------------------------------------------------------
__global__ __launch_bounds__(256) void transpose_cvt_kernel(const float* __restrict__ W,
                                                            f16* __restrict__ Wt,
                                                            int R, int C) {
  __shared__ float tile[32][33];
  const int c0 = blockIdx.x * 32, r0 = blockIdx.y * 32;
  const int tx = threadIdx.x, ty = threadIdx.y;   // 32 x 8
  for (int i = 0; i < 32; i += 8)
    tile[ty + i][tx] = W[(size_t)(r0 + ty + i) * C + c0 + tx];
  __syncthreads();
  for (int i = 0; i < 32; i += 8)
    Wt[(size_t)(c0 + ty + i) * R + r0 + tx] = (f16)tile[tx][ty + i];
}

// ---------------------------------------------------------------------------
// WMMA GEMM: C[M][N] = A[M][K] * Bt[N][K]^T   (both operands f16, f32 accum)
// Block tile 128x128, 8 waves (2M x 4N), per-wave 64x32 -> 4x2 accum tiles.
// K staged 32 at a time into DOUBLE-BUFFERED LDS via async-to-LDS loads:
// next tile's DMA is in flight while the current tile feeds WMMAs.
// Each thread issues 4 async b128 loads per tile, so s_wait_asynccnt 0x4
// means "current buffer complete, next buffer still loading".
// MODE 0: write f32 to Cout[M][N]
// MODE 1: QKV scatter -> Qh[b,h,s,d]*0.125, Kh[b,h,s,d], VTh[b,h,d,s] (f16)
// ---------------------------------------------------------------------------
template <int MODE>
__global__ __launch_bounds__(256)
void gemm_f16_kernel(const f16* __restrict__ A, const f16* __restrict__ Bt,
                     float* __restrict__ Cout,
                     f16* __restrict__ Qh, f16* __restrict__ Kh, f16* __restrict__ VTh,
                     int Ndim, int Kdim) {
  __shared__ f16 As[2][128 * 32];   // 8KB per buffer
  __shared__ f16 Bs[2][128 * 32];
  const int tid   = threadIdx.x;
  const int lane  = tid & 31;
  const int wave  = tid >> 5;
  const int waveM = wave >> 2;   // 0..1
  const int waveN = wave & 3;    // 0..3
  const int m0 = blockIdx.y * 128;
  const int n0 = blockIdx.x * 128;
  const int r = lane & 15, hi = lane >> 4;

  // Staging assignment: 128 rows x 4 16B-chunks per array = 512 chunks;
  // 256 threads -> 2 chunks per array per thread -> 4 async loads per tile.
  const int srow0 = tid >> 2,         sch0 = tid & 3;
  const int srow1 = (tid + 256) >> 2, sch1 = (tid + 256) & 3;

  v8f acc[4][2];
  for (int mi = 0; mi < 4; ++mi)
    for (int ni = 0; ni < 2; ++ni)
      for (int j = 0; j < 8; ++j) acc[mi][ni][j] = 0.f;

  // Prologue: start DMA of tile 0 into buffer 0.
  {
    const int k0 = 0;
    async_copy_b128(A  + (size_t)(m0 + srow0) * Kdim + k0 + sch0 * 8, &As[0][srow0 * 32 + sch0 * 8]);
    async_copy_b128(Bt + (size_t)(n0 + srow0) * Kdim + k0 + sch0 * 8, &Bs[0][srow0 * 32 + sch0 * 8]);
    async_copy_b128(A  + (size_t)(m0 + srow1) * Kdim + k0 + sch1 * 8, &As[0][srow1 * 32 + sch1 * 8]);
    async_copy_b128(Bt + (size_t)(n0 + srow1) * Kdim + k0 + sch1 * 8, &Bs[0][srow1 * 32 + sch1 * 8]);
  }

  const int nIter = Kdim >> 5;
  for (int it = 0; it < nIter; ++it) {
    const int buf = it & 1;
    const bool hasNext = (it + 1) < nIter;
    if (hasNext) {  // kick off next tile's DMA into the other buffer
      const int k1 = (it + 1) << 5;
      async_copy_b128(A  + (size_t)(m0 + srow0) * Kdim + k1 + sch0 * 8, &As[buf ^ 1][srow0 * 32 + sch0 * 8]);
      async_copy_b128(Bt + (size_t)(n0 + srow0) * Kdim + k1 + sch0 * 8, &Bs[buf ^ 1][srow0 * 32 + sch0 * 8]);
      async_copy_b128(A  + (size_t)(m0 + srow1) * Kdim + k1 + sch1 * 8, &As[buf ^ 1][srow1 * 32 + sch1 * 8]);
      async_copy_b128(Bt + (size_t)(n0 + srow1) * Kdim + k1 + sch1 * 8, &Bs[buf ^ 1][srow1 * 32 + sch1 * 8]);
      asm volatile("s_wait_asynccnt 0x4" ::: "memory");   // current buf done, next in flight
    } else {
      asm volatile("s_wait_asynccnt 0x0" ::: "memory");
    }
    __syncthreads();   // publish this buffer's LDS writes to all waves

    v16h av[4], bv[2];
    for (int mi = 0; mi < 4; ++mi)
      av[mi] = load_frag_a(&As[buf][(waveM * 64 + mi * 16) * 32], 32, lane);
    for (int ni = 0; ni < 2; ++ni)
      bv[ni] = load_frag_b(&Bs[buf][(waveN * 32 + ni * 16) * 32], 32, lane);

    for (int mi = 0; mi < 4; ++mi)
      for (int ni = 0; ni < 2; ++ni)
        acc[mi][ni] = __builtin_amdgcn_wmma_f32_16x16x32_f16(
            false, av[mi], false, bv[ni], (short)0, acc[mi][ni], false, false);
    __syncthreads();   // everyone done reading buf before it is re-filled
  }

  // Writeback. C-layout: element (M = j + 8*hi, N = r) within each 16x16 tile.
  for (int mi = 0; mi < 4; ++mi)
    for (int ni = 0; ni < 2; ++ni)
      for (int j = 0; j < 8; ++j) {
        int gm = m0 + waveM * 64 + mi * 16 + j + 8 * hi;
        int gn = n0 + waveN * 32 + ni * 16 + r;
        float v = acc[mi][ni][j];
        if (MODE == 0) {
          Cout[(size_t)gm * Ndim + gn] = v;
        } else {
          int b   = gm >> 11;          // / S_
          int s   = gm & (S_ - 1);
          int sec = gn >> 10;          // 0=Q 1=K 2=V
          int hc  = gn & (D_ - 1);
          int h   = hc >> 6;
          int d   = hc & 63;
          size_t bh = (size_t)b * H_ + h;
          if (sec == 0)      Qh[(bh * S_ + s) * HD_ + d] = (f16)(v * 0.125f); // 1/sqrt(64)
          else if (sec == 1) Kh[(bh * S_ + s) * HD_ + d] = (f16)v;
          else               VTh[(bh * HD_ + d) * S_ + s] = (f16)v;           // V transposed
        }
      }
}

// ---------------------------------------------------------------------------
// Flash-attention (causal, online softmax). Grid: (S/64, B*H), 4 waves/block.
// Each wave owns 16 query rows; all WMMA 16x16x32 f16, f32 accumulation.
// Q pre-scaled by 1/sqrt(HD). K rows and VT rows are already in Bt layout.
// P (C-layout) -> A-layout via per-wave LDS round trip (no cross-wave deps,
// so only s_wait_dscnt is needed, no barrier; all waves run the same kv trip
// count).
// ---------------------------------------------------------------------------
__global__ __launch_bounds__(128)
void attn_kernel(const f16* __restrict__ Qh, const f16* __restrict__ Kh,
                 const f16* __restrict__ VTh, f16* __restrict__ Yh) {
  __shared__ f16 Ps[4][16 * 64];   // per-wave P tile, 2KB each
  const int tid  = threadIdx.x;
  const int lane = tid & 31;
  const int wave = tid >> 5;
  const int r = lane & 15, hi = lane >> 4;
  const int q0 = blockIdx.x * 64;
  const int bh = blockIdx.y;
  const int b  = bh >> 4;          // H_=16
  const int h  = bh & 15;
  const int qw = q0 + wave * 16;   // this wave's 16 query rows

  const f16* qbase = Qh  + ((size_t)bh * S_ + qw) * HD_;
  const f16* kbase = Kh  + (size_t)bh * S_ * HD_;
  const f16* vbase = VTh + (size_t)bh * HD_ * S_;

  const v16h qf0 = load_frag_a(qbase + 0,  HD_, lane);
  const v16h qf1 = load_frag_a(qbase + 32, HD_, lane);

  v8f O[4];
  float mrow[8], lrow[8];
  for (int dt = 0; dt < 4; ++dt)
    for (int j = 0; j < 8; ++j) O[dt][j] = 0.f;
  for (int j = 0; j < 8; ++j) { mrow[j] = -1e30f; lrow[j] = 0.f; }

  f16* pw = &Ps[wave][0];

  for (int kv0 = 0; kv0 <= qw + 15; kv0 += 64) {
    // S = Q * K^T  (4 key tiles x 2 K-steps over head dim)
    v8f sc4[4];
    for (int ni = 0; ni < 4; ++ni)
      for (int j = 0; j < 8; ++j) sc4[ni][j] = 0.f;
    for (int ni = 0; ni < 4; ++ni) {
      v16h kb0 = load_frag_b(kbase + (size_t)(kv0 + ni * 16) * HD_ + 0,  HD_, lane);
      v16h kb1 = load_frag_b(kbase + (size_t)(kv0 + ni * 16) * HD_ + 32, HD_, lane);
      sc4[ni] = __builtin_amdgcn_wmma_f32_16x16x32_f16(false, qf0, false, kb0, (short)0, sc4[ni], false, false);
      sc4[ni] = __builtin_amdgcn_wmma_f32_16x16x32_f16(false, qf1, false, kb1, (short)0, sc4[ni], false, false);
    }
    // Causal mask (only tiles touching/above the diagonal)
    if (kv0 + 63 > qw) {
      for (int ni = 0; ni < 4; ++ni)
        for (int j = 0; j < 8; ++j) {
          int key = kv0 + ni * 16 + r;
          int q   = qw + j + 8 * hi;
          if (key > q) sc4[ni][j] = -1e30f;
        }
    }
    // Online softmax per row (row M spread across 16 lanes, xor-reduce)
    for (int j = 0; j < 8; ++j) {
      float rm = -1e30f;
      for (int ni = 0; ni < 4; ++ni) rm = fmaxf(rm, sc4[ni][j]);
      for (int off = 1; off < 16; off <<= 1) rm = fmaxf(rm, __shfl_xor(rm, off));
      float mn    = fmaxf(mrow[j], rm);
      float scale = __expf(mrow[j] - mn);
      float ps = 0.f;
      for (int ni = 0; ni < 4; ++ni) {
        float p = __expf(sc4[ni][j] - mn);
        sc4[ni][j] = p;
        ps += p;
      }
      for (int off = 1; off < 16; off <<= 1) ps += __shfl_xor(ps, off);
      lrow[j] = lrow[j] * scale + ps;
      mrow[j] = mn;
      for (int dt = 0; dt < 4; ++dt) O[dt][j] *= scale;
      // spill P row elements to LDS in [16][64] row-major (f16)
      for (int ni = 0; ni < 4; ++ni)
        pw[(j + 8 * hi) * 64 + ni * 16 + r] = (f16)sc4[ni][j];
    }
    // Wave-local LDS RAW fence (cross-lane, compiler can't see it)
    asm volatile("s_wait_dscnt 0x0" ::: "memory");

    // O += P * V   (P as A-matrix from LDS; VT rows are Bt layout)
    v16h pf0 = load_frag_a(pw + 0,  64, lane);
    v16h pf1 = load_frag_a(pw + 32, 64, lane);
    for (int dt = 0; dt < 4; ++dt) {
      v16h vb0 = load_frag_b(vbase + (size_t)(dt * 16) * S_ + kv0 + 0,  S_, lane);
      v16h vb1 = load_frag_b(vbase + (size_t)(dt * 16) * S_ + kv0 + 32, S_, lane);
      O[dt] = __builtin_amdgcn_wmma_f32_16x16x32_f16(false, pf0, false, vb0, (short)0, O[dt], false, false);
      O[dt] = __builtin_amdgcn_wmma_f32_16x16x32_f16(false, pf1, false, vb1, (short)0, O[dt], false, false);
    }
  }

  // Normalize and write Y[b][s][h*64 + d] as f16 (input to proj GEMM)
  for (int dt = 0; dt < 4; ++dt)
    for (int j = 0; j < 8; ++j) {
      float y = O[dt][j] / lrow[j];
      size_t row = (size_t)b * S_ + qw + j + 8 * hi;
      Yh[row * D_ + h * HD_ + dt * 16 + r] = (f16)y;
    }
}

// ---------------------------------------------------------------------------
// Host launch
// ---------------------------------------------------------------------------
extern "C" void kernel_launch(void* const* d_in, const int* in_sizes, int n_in,
                              void* d_out, int out_size, void* d_ws, size_t ws_size,
                              hipStream_t stream) {
  (void)in_sizes; (void)n_in; (void)out_size; (void)ws_size;
  const float* x     = (const float*)d_in[0];   // [4,2048,1024]
  const float* Wqkv  = (const float*)d_in[1];   // [1024,3072]
  const float* Wproj = (const float*)d_in[2];   // [1024,1024]
  float* out = (float*)d_out;                   // [4,2048,1024]

  // Workspace layout (f16)
  f16* xh     = (f16*)d_ws;                       // [8192][1024]
  f16* wqkvT  = xh     + (size_t)MROWS * D_;      // [3072][1024]
  f16* wprojT = wqkvT  + (size_t)NQKV * D_;       // [1024][1024]
  f16* Qh     = wprojT + (size_t)D_ * D_;         // [B,H,S,64] (pre-scaled)
  f16* Kh     = Qh     + (size_t)B_ * H_ * S_ * HD_;
  f16* VT     = Kh     + (size_t)B_ * H_ * S_ * HD_;   // [B,H,64,S]
  f16* Yh     = VT     + (size_t)B_ * H_ * S_ * HD_;   // [8192][1024]

  const int nX = MROWS * D_;   // 8,388,608
  cvt_f32_f16_kernel<<<nX / (256 * 4), 256, 0, stream>>>(x, xh, nX);

  transpose_cvt_kernel<<<dim3(NQKV / 32, D_ / 32), dim3(32, 8), 0, stream>>>(Wqkv, wqkvT, D_, NQKV);
  transpose_cvt_kernel<<<dim3(D_ / 32, D_ / 32),  dim3(32, 8), 0, stream>>>(Wproj, wprojT, D_, D_);

  // QKV projection, scattering straight into attention layouts
  gemm_f16_kernel<1><<<dim3(NQKV / 128, MROWS / 128), 256, 0, stream>>>(
      xh, wqkvT, nullptr, Qh, Kh, VT, NQKV, D_);

  // Causal flash attention
  attn_kernel<<<dim3(S_ / 64, B_ * H_), 128, 0, stream>>>(Qh, Kh, VT, Yh);

  // Output projection -> f32 result
  gemm_f16_kernel<0><<<dim3(D_ / 128, MROWS / 128), 256, 0, stream>>>(
      Yh, wprojT, out, nullptr, nullptr, nullptr, D_, D_);
}
